// HierarchicalPointCloudProcessor_70179765617263
// MI455X (gfx1250) — matrix operand order repaired and checked by reading, not compiled, not run on hardware
//
#include <hip/hip_runtime.h>
#include <hip/hip_bf16.h>
#include <math.h>

// ---------------------------------------------------------------------------
// Hierarchical point-cloud GNN for MI455X (gfx1250, wave32).
// All dense matmuls run through v_wmma_f32_16x16x32_f16 with pre-packed,
// pre-padded f16 operands: A row-major [M, Kp], B transposed [N, Kp].
// Each wave computes a 16x64 C tile (NT=4 accumulators) so the A fragment is
// reused 4x and independent WMMAs pipeline back-to-back; every fragment is
// two aligned 16B global_load_b128 loads, zero guards on the hot path.
// Scatter-softmax uses global_atomic_max (sortable-uint) + atomic_add_f32.
// ---------------------------------------------------------------------------

typedef __attribute__((ext_vector_type(16))) _Float16 v16h;
typedef __attribute__((ext_vector_type(8)))  _Float16 v8h;
typedef __attribute__((ext_vector_type(8)))  float    v8f;

#define K_NN 20
#define CEILDIV(a, b) (((a) + (b) - 1) / (b))

__device__ __forceinline__ unsigned fenc(float f) {
    unsigned u = __float_as_uint(f);
    return (u & 0x80000000u) ? ~u : (u | 0x80000000u);
}
__device__ __forceinline__ float fdec(unsigned x) {
    return __uint_as_float((x & 0x80000000u) ? (x ^ 0x80000000u) : ~x);
}
__device__ __forceinline__ float gelu_erf(float v) {
    return 0.5f * v * (1.f + erff(v * 0.7071067811865475f));
}

// ---------------------------------------------------------------------------
// WMMA GEMM, guard-free: C = act(A * B^T_packed + bias)
//   A  : f16 [M, KT*32] row-major (M % 16 == 0, fully padded)
//   BT : f16 [N, KT*32] row-major (= B transposed, N % (16*NT) == 0, padded)
//   C32/C16: optional f32 / f16 outputs (either may be null)
// One wave per 16x(16*NT) tile; grid = (M/16, N/(16*NT)). ACT: 0=none, 1=relu.
// ---------------------------------------------------------------------------
template <int KT, int NT, int ACT>
__global__ void wmma_gemm_kernel(const _Float16* __restrict__ A, int lda,
                                 const _Float16* __restrict__ BT, int ldbt,
                                 const float* __restrict__ bias,
                                 float* __restrict__ C32, int ldc32,
                                 _Float16* __restrict__ C16, int ldc16) {
    const int lane = threadIdx.x & 31;
    const int half = lane >> 4;
    const int l15 = lane & 15;
    const int arow = blockIdx.x * 16 + l15;
    const int bcol0 = blockIdx.y * (16 * NT) + l15;

    // A fragment: lanes 0-15 K-base 0, lanes 16-31 K-base 8; halves 8..15 at +16
    const _Float16* Ar = A + (size_t)arow * lda + (half ? 8 : 0);
    // B fragment: lanes 0-15 K-base 0, lanes 16-31 K-base 8; halves 8..15 at +16
    const _Float16* Br0 = BT + (size_t)bcol0 * ldbt + (half ? 8 : 0);

    v8f c[NT];
#pragma unroll
    for (int nt = 0; nt < NT; ++nt) {
        const float cb = (bias != nullptr) ? bias[bcol0 + nt * 16] : 0.f;
#pragma unroll
        for (int v = 0; v < 8; ++v) c[nt][v] = cb;
    }

#pragma unroll
    for (int kt = 0; kt < KT; ++kt) {
        v8h a0 = *(const v8h*)(Ar + kt * 32);
        v8h a1 = *(const v8h*)(Ar + kt * 32 + 16);
        v16h a = __builtin_shufflevector(a0, a1, 0, 1, 2, 3, 4, 5, 6, 7,
                                         8, 9, 10, 11, 12, 13, 14, 15);
#pragma unroll
        for (int nt = 0; nt < NT; ++nt) {
            const _Float16* Br = Br0 + (size_t)nt * 16 * ldbt;
            v8h b0 = *(const v8h*)(Br + kt * 32);
            v8h b1 = *(const v8h*)(Br + kt * 32 + 16);
            v16h b = __builtin_shufflevector(b0, b1, 0, 1, 2, 3, 4, 5, 6, 7,
                                             8, 9, 10, 11, 12, 13, 14, 15);
            c[nt] = __builtin_amdgcn_wmma_f32_16x16x32_f16(false, a, false, b,
                                                           (short)0, c[nt], false, false);
        }
    }

    const int rbase = blockIdx.x * 16 + (half ? 8 : 0);
    if (C32 != nullptr) {
#pragma unroll
        for (int nt = 0; nt < NT; ++nt) {
            const int col = bcol0 + nt * 16;
#pragma unroll
            for (int v = 0; v < 8; ++v) {
                float x = c[nt][v];
                if (ACT == 1) x = fmaxf(x, 0.f);
                C32[(size_t)(rbase + v) * ldc32 + col] = x;
            }
        }
    }
    if (C16 != nullptr) {
#pragma unroll
        for (int nt = 0; nt < NT; ++nt) {
            const int col = bcol0 + nt * 16;
#pragma unroll
            for (int v = 0; v < 8; ++v) {
                float x = c[nt][v];
                if (ACT == 1) x = fmaxf(x, 0.f);
                C16[(size_t)(rbase + v) * ldc16 + col] = (_Float16)x;
            }
        }
    }
}

// pack f32 [Msrc, K] row-major -> f16 [Mp, Kp] row-major, zero padded
__global__ void pack_a_f16_kernel(const float* __restrict__ src, int Msrc, int K,
                                  _Float16* __restrict__ dst, int Mp, int Kp) {
    int t = blockIdx.x * blockDim.x + threadIdx.x;
    if (t >= Mp * Kp) return;
    int m = t / Kp, k = t % Kp;
    dst[t] = (m < Msrc && k < K) ? (_Float16)src[(size_t)m * K + k] : (_Float16)0.f;
}

// pack f32 B [K, N] -> f16 B^T [Np, Kp], zero padded
__global__ void pack_bt_f16_kernel(const float* __restrict__ B, int K, int N,
                                   _Float16* __restrict__ BT, int Kp, int Np) {
    int t = blockIdx.x * blockDim.x + threadIdx.x;
    if (t >= Np * Kp) return;
    int n = t / Kp, k = t % Kp;
    BT[t] = (k < K && n < N) ? (_Float16)B[(size_t)k * N + n] : (_Float16)0.f;
}

// ---------------------------------------------------------------------------
// Brute force kNN (k=20): one query per thread, LDS-tiled candidates,
// register insertion-sorted top-20 (ascending squared distance).
// ---------------------------------------------------------------------------
template <int DIM, bool INCSELF>
__global__ void knn_kernel(const float* __restrict__ P, int n, int* __restrict__ out_idx) {
    const int TILE = 128;
    __shared__ float sP[TILE * DIM];
    int q = blockIdx.x * blockDim.x + threadIdx.x;
    float qv[DIM];
#pragma unroll
    for (int d = 0; d < DIM; ++d) qv[d] = (q < n) ? P[(size_t)q * DIM + d] : 0.f;
    float bd[K_NN];
    int bi[K_NN];
#pragma unroll
    for (int t = 0; t < K_NN; ++t) { bd[t] = INFINITY; bi[t] = 0; }

    for (int base = 0; base < n; base += TILE) {
        __syncthreads();
        for (int t = threadIdx.x; t < TILE * DIM; t += blockDim.x) {
            int r = base + t / DIM;
            sP[t] = (r < n) ? P[(size_t)r * DIM + (t % DIM)] : 0.f;
        }
        __syncthreads();
        int lim = min(TILE, n - base);
        for (int j = 0; j < lim; ++j) {
            int cj = base + j;
            float d2 = 0.f;
#pragma unroll
            for (int d = 0; d < DIM; ++d) {
                float df = qv[d] - sP[j * DIM + d];
                d2 = fmaf(df, df, d2);
            }
            if (!INCSELF && cj == q) d2 = INFINITY;  // branchless self-exclusion
            if (d2 < bd[K_NN - 1]) {
                int p = K_NN - 1;
                while (p > 0 && d2 < bd[p - 1]) {
                    bd[p] = bd[p - 1];
                    bi[p] = bi[p - 1];
                    --p;
                }
                bd[p] = d2;
                bi[p] = cj;
            }
        }
    }
    if (q < n) {
#pragma unroll
        for (int t = 0; t < K_NN; ++t) out_idx[(size_t)q * K_NN + t] = bi[t];
    }
}

// xp = [x || pos]  (8 + 3 = 11)
__global__ void concat_xp_kernel(const float* __restrict__ x, const float* __restrict__ pos,
                                 float* __restrict__ xp, int n) {
    int t = blockIdx.x * blockDim.x + threadIdx.x;
    if (t >= n) return;
#pragma unroll
    for (int d = 0; d < 8; ++d) xp[(size_t)t * 11 + d] = x[(size_t)t * 8 + d];
#pragma unroll
    for (int d = 0; d < 3; ++d) xp[(size_t)t * 11 + 8 + d] = pos[(size_t)t * 3 + d];
}

// edge features m16[e] = f16 [xp_i (11) || xp_j - xp_i (11) || 0-pad (10)]
__global__ void edge_gather1_kernel(const float* __restrict__ xp, const int* __restrict__ idx,
                                    _Float16* __restrict__ m16, int n) {
    int e = blockIdx.x * blockDim.x + threadIdx.x;
    if (e >= n * K_NN) return;
    int i = e / K_NN;
    int j = idx[e];
    _Float16* r = m16 + (size_t)e * 32;
#pragma unroll
    for (int d = 0; d < 11; ++d) {
        float a = xp[(size_t)i * 11 + d];
        float b = xp[(size_t)j * 11 + d];
        r[d] = (_Float16)a;
        r[11 + d] = (_Float16)(b - a);
    }
#pragma unroll
    for (int d = 22; d < 32; ++d) r[d] = (_Float16)0.f;
}

// per-point channel-wise max over K_NN edges
__global__ void max_over_k_kernel(const float* __restrict__ H, float* __restrict__ out,
                                  int n, int C) {
    int t = blockIdx.x * blockDim.x + threadIdx.x;
    if (t >= n * C) return;
    int i = t / C, c = t % C;
    const float* base = H + (size_t)i * K_NN * C + c;
    float mx = base[0];
#pragma unroll
    for (int k = 1; k < K_NN; ++k) mx = fmaxf(mx, base[(size_t)k * C]);
    out[(size_t)i * C + c] = mx;
}

// LayerNorm (eps 1e-5) + exact-erf GELU, one wave per row
__global__ void ln_gelu_kernel(const float* __restrict__ X, const float* __restrict__ g,
                               const float* __restrict__ b, float* __restrict__ out,
                               int n, int C) {
    int row = blockIdx.x * (blockDim.x >> 5) + (threadIdx.x >> 5);
    int lane = threadIdx.x & 31;
    if (row >= n) return;
    const float* x = X + (size_t)row * C;
    float s = 0.f, s2 = 0.f;
    for (int c = lane; c < C; c += 32) {
        float v = x[c];
        s += v;
        s2 = fmaf(v, v, s2);
    }
#pragma unroll
    for (int off = 16; off > 0; off >>= 1) {
        s += __shfl_xor(s, off, 32);
        s2 += __shfl_xor(s2, off, 32);
    }
    float mu = s / C;
    float inv = rsqrtf(s2 / C - mu * mu + 1e-5f);
    for (int c = lane; c < C; c += 32) {
        float v = (x[c] - mu) * inv * g[c] + b[c];
        out[(size_t)row * C + c] = gelu_erf(v);
    }
}

// s = sigmoid(relu([feat||pos] @ w1 + b1) @ w2 + b2); one wave per row
__global__ void score_kernel(const float* __restrict__ F, const float* __restrict__ P,
                             int Fd, int H,
                             const float* __restrict__ w1, const float* __restrict__ b1,
                             const float* __restrict__ w2, const float* __restrict__ b2,
                             float* __restrict__ s, int n) {
    int row = blockIdx.x * (blockDim.x >> 5) + (threadIdx.x >> 5);
    int lane = threadIdx.x & 31;
    if (row >= n) return;
    float acc = 0.f;
    for (int h = lane; h < H; h += 32) {
        float hv = b1[h];
        for (int d = 0; d < Fd; ++d) hv = fmaf(F[(size_t)row * Fd + d], w1[(size_t)d * H + h], hv);
#pragma unroll
        for (int d = 0; d < 3; ++d) hv = fmaf(P[(size_t)row * 3 + d], w1[(size_t)(Fd + d) * H + h], hv);
        hv = fmaxf(hv, 0.f);
        acc = fmaf(hv, w2[h], acc);
    }
#pragma unroll
    for (int off = 16; off > 0; off >>= 1) acc += __shfl_xor(acc, off, 32);
    if (lane == 0) s[row] = 1.f / (1.f + expf(-(acc + b2[0])));
}

// exact stable top-k via O(n^2) ranking (ties -> lower index, matches lax.top_k)
__global__ void rank_select_kernel(const float* __restrict__ s, int n, int m,
                                   int* __restrict__ sel) {
    __shared__ float sv[256];
    int i = blockIdx.x * blockDim.x + threadIdx.x;
    float si = (i < n) ? s[i] : 0.f;
    int rank = 0;
    for (int base = 0; base < n; base += 256) {
        __syncthreads();
        sv[threadIdx.x] = (base + (int)threadIdx.x < n) ? s[base + threadIdx.x] : -INFINITY;
        __syncthreads();
        if (i < n) {
            for (int j = 0; j < 256; ++j) {
                int cj = base + j;
                float sj = sv[j];
                if (sj > si || (sj == si && cj < i)) ++rank;
            }
        }
    }
    if (i < n && rank < m) sel[rank] = i;
}

__global__ void gather_rows_kernel(const float* __restrict__ src, const int* __restrict__ sel,
                                   float* __restrict__ dst, int m, int C) {
    int t = blockIdx.x * blockDim.x + threadIdx.x;
    if (t >= m * C) return;
    int r = t / C, c = t % C;
    dst[(size_t)r * C + c] = src[(size_t)sel[r] * C + c];
}

__global__ void mean_rows_kernel(const float* __restrict__ X, float* __restrict__ out,
                                 int n, int C) {
    int c = blockIdx.x * blockDim.x + threadIdx.x;
    if (c >= C) return;
    float s = 0.f;
    for (int r = 0; r < n; ++r) s += X[(size_t)r * C + c];
    out[c] = s / (float)n;
}

// level-2 edge geometry -> f16 padded [E2, 32]
__global__ void edge_dpos_kernel(const float* __restrict__ pos2, const int* __restrict__ nbr,
                                 _Float16* __restrict__ dpos16, int n2) {
    int e = blockIdx.x * blockDim.x + threadIdx.x;
    if (e >= n2 * K_NN) return;
    int src = e / K_NN;
    int tgt = nbr[e];
    _Float16* r = dpos16 + (size_t)e * 32;
#pragma unroll
    for (int d = 0; d < 3; ++d)
        r[d] = (_Float16)(pos2[(size_t)tgt * 3 + d] - pos2[(size_t)src * 3 + d]);
#pragma unroll
    for (int d = 3; d < 32; ++d) r[d] = (_Float16)0.f;
}

// ain16 = f16( xd[tgt] - xs[src] + delta )   (per edge, 256 channels)
__global__ void alpha_in_kernel(const float* __restrict__ xd, const float* __restrict__ xs,
                                const float* __restrict__ delta, const int* __restrict__ nbr,
                                _Float16* __restrict__ ain16, int E) {
    int t = blockIdx.x * blockDim.x + threadIdx.x;
    if (t >= E * 256) return;
    int e = t >> 8, c = t & 255;
    int src = e / K_NN, tgt = nbr[e];
    ain16[t] = (_Float16)(xd[(size_t)tgt * 256 + c] - xs[(size_t)src * 256 + c] + delta[t]);
}

__global__ void fill_zero_kernel(unsigned* __restrict__ p, int n) {
    int t = blockIdx.x * blockDim.x + threadIdx.x;
    if (t < n) p[t] = 0u;
}

__global__ void scatter_max_kernel(const float* __restrict__ alpha, const int* __restrict__ nbr,
                                   unsigned* __restrict__ amax, int E) {
    int t = blockIdx.x * blockDim.x + threadIdx.x;
    if (t >= E * 256) return;
    int e = t >> 8, c = t & 255;
    int tgt = nbr[e];
    atomicMax(&amax[(size_t)tgt * 256 + c], fenc(alpha[t]));
}

__global__ void scatter_sum_kernel(const float* __restrict__ alpha, const float* __restrict__ delta,
                                   const float* __restrict__ xl, const int* __restrict__ nbr,
                                   const unsigned* __restrict__ amax,
                                   float* __restrict__ den, float* __restrict__ num, int E) {
    int t = blockIdx.x * blockDim.x + threadIdx.x;
    if (t >= E * 256) return;
    int e = t >> 8, c = t & 255;
    int src = e / K_NN;
    int tgt = nbr[e];
    float am = fdec(amax[(size_t)tgt * 256 + c]);
    float ev = expf(alpha[t] - am);
    atomicAdd(&den[(size_t)tgt * 256 + c], ev);
    atomicAdd(&num[(size_t)tgt * 256 + c], ev * (xl[(size_t)src * 256 + c] + delta[t]));
}

__global__ void softmax_div_kernel(const float* __restrict__ num, const float* __restrict__ den,
                                   float* __restrict__ out, int n) {
    int t = blockIdx.x * blockDim.x + threadIdx.x;
    if (t < n) out[t] = num[t] / (den[t] + 1e-16f);
}

// s3 strip: s3[r][0..3] = s3p[r][0..3] + gn_s_b  (undo N=16 padding)
__global__ void strip_s3_kernel(const float* __restrict__ s3p, const float* __restrict__ bias,
                                float* __restrict__ s3, int n3) {
    int t = blockIdx.x * blockDim.x + threadIdx.x;
    if (t >= n3 * 4) return;
    int r = t >> 2, d = t & 3;
    s3[t] = s3p[(size_t)r * 16 + d] + bias[d];
}

// GravNet messages -> f16 gin [n3, 288] = [x3 (256) || mean_k (16) || max_k (16)]
__global__ void gravnet_kernel(const float* __restrict__ x3, const float* __restrict__ s3,
                               const float* __restrict__ h3, const int* __restrict__ idx3,
                               _Float16* __restrict__ gin16, int n3) {
    int row = blockIdx.x * (blockDim.x >> 5) + (threadIdx.x >> 5);
    int lane = threadIdx.x & 31;
    if (row >= n3) return;
    for (int c = lane; c < 256; c += 32)
        gin16[(size_t)row * 288 + c] = (_Float16)x3[(size_t)row * 256 + c];
    if (lane < 16) {
        float q0 = s3[row * 4 + 0], q1 = s3[row * 4 + 1];
        float q2 = s3[row * 4 + 2], q3 = s3[row * 4 + 3];
        float mean = 0.f, mx = -INFINITY;
#pragma unroll
        for (int k = 0; k < K_NN; ++k) {
            int j = idx3[row * K_NN + k];
            float d0 = q0 - s3[j * 4 + 0];
            float d1 = q1 - s3[j * 4 + 1];
            float d2 = q2 - s3[j * 4 + 2];
            float d3 = q3 - s3[j * 4 + 3];
            float dd = d0 * d0 + d1 * d1 + d2 * d2 + d3 * d3;
            float wv = expf(-10.f * dd);
            float mv = h3[(size_t)j * 16 + lane] * wv;
            mean += mv;
            mx = fmaxf(mx, mv);
        }
        gin16[(size_t)row * 288 + 256 + lane] = (_Float16)(mean * (1.f / (float)K_NN));
        gin16[(size_t)row * 288 + 272 + lane] = (_Float16)mx;
    }
}

// ---------------------------------------------------------------------------
// Orchestration
// ---------------------------------------------------------------------------
extern "C" void kernel_launch(void* const* d_in, const int* in_sizes, int n_in,
                              void* d_out, int out_size, void* d_ws, size_t ws_size,
                              hipStream_t stream) {
    (void)in_sizes; (void)n_in; (void)out_size; (void)ws_size;

    const float* x       = (const float*)d_in[0];
    const float* pos     = (const float*)d_in[1];
    const float* dec_w1  = (const float*)d_in[2];
    const float* dec_b1  = (const float*)d_in[3];
    const float* dec_w2  = (const float*)d_in[4];
    const float* dec_b2  = (const float*)d_in[5];
    const float* dec_w3  = (const float*)d_in[6];
    const float* dec_b3  = (const float*)d_in[7];
    const float* ln1_g   = (const float*)d_in[8];
    const float* ln1_b   = (const float*)d_in[9];
    const float* ds1_w1  = (const float*)d_in[10];
    const float* ds1_b1  = (const float*)d_in[11];
    const float* ds1_w2  = (const float*)d_in[12];
    const float* ds1_b2  = (const float*)d_in[13];
    const float* pt_lin  = (const float*)d_in[14];
    const float* pt_src  = (const float*)d_in[15];
    const float* pt_dst  = (const float*)d_in[16];
    const float* pt_pw1  = (const float*)d_in[17];
    const float* pt_pb1  = (const float*)d_in[18];
    const float* pt_pw2  = (const float*)d_in[19];
    const float* pt_pb2  = (const float*)d_in[20];
    const float* pt_aw1  = (const float*)d_in[21];
    const float* pt_ab1  = (const float*)d_in[22];
    const float* pt_aw2  = (const float*)d_in[23];
    const float* pt_ab2  = (const float*)d_in[24];
    const float* ln2_g   = (const float*)d_in[25];
    const float* ln2_b   = (const float*)d_in[26];
    const float* ds2_w1  = (const float*)d_in[27];
    const float* ds2_b1  = (const float*)d_in[28];
    const float* ds2_w2  = (const float*)d_in[29];
    const float* ds2_b2  = (const float*)d_in[30];
    const float* gn_s_w  = (const float*)d_in[31];
    const float* gn_s_b  = (const float*)d_in[32];
    const float* gn_h_w  = (const float*)d_in[33];
    const float* gn_h_b  = (const float*)d_in[34];
    const float* gn_o_w  = (const float*)d_in[35];
    const float* gn_o_b  = (const float*)d_in[36];
    const float* ln3_g   = (const float*)d_in[37];
    const float* ln3_b   = (const float*)d_in[38];
    const float* fin_w   = (const float*)d_in[39];
    const float* fin_b   = (const float*)d_in[40];
    const float* fln_g   = (const float*)d_in[41];
    const float* fln_b   = (const float*)d_in[42];

    const int n1 = 8192, n2 = 4096, n3 = 1024;
    const int E1 = n1 * K_NN;   // 163840
    const int E2 = n2 * K_NN;   // 81920

    float* outp = (float*)d_out;
    float* o_out  = outp + 0;              // [1,1024]
    float* o_f1   = outp + 1024;           // [8192,128]
    float* o_pos  = outp + 1049600;        // [8192,3]
    float* o_f2   = outp + 1074176;        // [4096,256]
    float* o_pos2 = outp + 2122752;        // [4096,3]
    float* o_f3   = outp + 2135040;        // [1024,512]
    float* o_pos3 = outp + 2659328;        // [1024,3]

    // ---- workspace bump allocator (256B aligned) ----
    char* wsb = (char*)d_ws;
    size_t off = 0;
    auto alloc = [&](size_t bytes) -> char* {
        char* p = wsb + off;
        off += (bytes + 255) & ~(size_t)255;
        return p;
    };
    // persistent f32 / int
    float*    xp   = (float*)alloc((size_t)n1 * 11 * 4);
    int*      idx1 = (int*)alloc((size_t)E1 * 4);
    float*    s1   = (float*)alloc((size_t)n1 * 4);
    int*      sel1 = (int*)alloc((size_t)n2 * 4);
    float*    x2   = (float*)alloc((size_t)n2 * 128 * 4);
    _Float16* x2h  = (_Float16*)alloc((size_t)n2 * 128 * 2);
    int*      nbr  = (int*)alloc((size_t)E2 * 4);
    float*    xl   = (float*)alloc((size_t)n2 * 256 * 4);
    float*    xs   = (float*)alloc((size_t)n2 * 256 * 4);
    float*    xd   = (float*)alloc((size_t)n2 * 256 * 4);
    unsigned* amax = (unsigned*)alloc((size_t)n2 * 256 * 4);  // contiguous with den,num
    float*    den  = (float*)alloc((size_t)n2 * 256 * 4);
    float*    num  = (float*)alloc((size_t)n2 * 256 * 4);
    float*    out2 = (float*)alloc((size_t)n2 * 256 * 4);
    float*    s2   = (float*)alloc((size_t)n2 * 4);
    int*      sel2 = (int*)alloc((size_t)n3 * 4);
    float*    x3   = (float*)alloc((size_t)n3 * 256 * 4);
    _Float16* x3h  = (_Float16*)alloc((size_t)n3 * 256 * 2);
    float*    s3p  = (float*)alloc((size_t)n3 * 16 * 4);
    float*    s3   = (float*)alloc((size_t)n3 * 4 * 4);
    float*    h3g  = (float*)alloc((size_t)n3 * 16 * 4);
    int*      idx3 = (int*)alloc((size_t)n3 * K_NN * 4);
    _Float16* gin16= (_Float16*)alloc((size_t)n3 * 288 * 2);
    float*    o3   = (float*)alloc((size_t)n3 * 512 * 4);
    float*    comb = (float*)alloc(896 * 4);
    _Float16* combh= (_Float16*)alloc((size_t)16 * 896 * 2);
    float*    ftmp = (float*)alloc((size_t)16 * 1024 * 4);
    // packed f16 weight transposes [N(p), Kp]
    _Float16* w1t  = (_Float16*)alloc((size_t)64 * 32 * 2);
    _Float16* w2t  = (_Float16*)alloc((size_t)128 * 64 * 2);
    _Float16* w3t  = (_Float16*)alloc((size_t)128 * 128 * 2);
    _Float16* plt  = (_Float16*)alloc((size_t)256 * 128 * 2);
    _Float16* pst  = (_Float16*)alloc((size_t)256 * 128 * 2);
    _Float16* pdt  = (_Float16*)alloc((size_t)256 * 128 * 2);
    _Float16* pw1t = (_Float16*)alloc((size_t)64 * 32 * 2);
    _Float16* pw2t = (_Float16*)alloc((size_t)256 * 64 * 2);
    _Float16* aw1t = (_Float16*)alloc((size_t)64 * 256 * 2);
    _Float16* aw2t = (_Float16*)alloc((size_t)256 * 64 * 2);
    _Float16* gst  = (_Float16*)alloc((size_t)16 * 256 * 2);
    _Float16* ght  = (_Float16*)alloc((size_t)16 * 256 * 2);
    _Float16* got  = (_Float16*)alloc((size_t)512 * 288 * 2);
    _Float16* fwt  = (_Float16*)alloc((size_t)1024 * 896 * 2);
    // big reusable arena
    float*    AR0  = (float*)alloc((size_t)E1 * 128 * 4);     // h3e | delta
    float*    AR1  = (float*)alloc((size_t)E2 * 256 * 4);     // alpha
    _Float16* AR2h = (_Float16*)alloc((size_t)E1 * 128 * 2);  // h2_16 | t1(f32) | ain16
    _Float16* AR3h = (_Float16*)alloc((size_t)E1 * 64 * 2);   // h1_16 | hh16 + hh2_16
    _Float16* AR4h = (_Float16*)alloc((size_t)E1 * 32 * 2);   // m16 | dpos16

    // ---- pack weight transposes (f16, padded) ----
    pack_bt_f16_kernel<<<CEILDIV(64 * 32, 256), 256, 0, stream>>>(dec_w1, 22, 64, w1t, 32, 64);
    pack_bt_f16_kernel<<<CEILDIV(128 * 64, 256), 256, 0, stream>>>(dec_w2, 64, 128, w2t, 64, 128);
    pack_bt_f16_kernel<<<CEILDIV(128 * 128, 256), 256, 0, stream>>>(dec_w3, 128, 128, w3t, 128, 128);
    pack_bt_f16_kernel<<<CEILDIV(256 * 128, 256), 256, 0, stream>>>(pt_lin, 128, 256, plt, 128, 256);
    pack_bt_f16_kernel<<<CEILDIV(256 * 128, 256), 256, 0, stream>>>(pt_src, 128, 256, pst, 128, 256);
    pack_bt_f16_kernel<<<CEILDIV(256 * 128, 256), 256, 0, stream>>>(pt_dst, 128, 256, pdt, 128, 256);
    pack_bt_f16_kernel<<<CEILDIV(64 * 32, 256), 256, 0, stream>>>(pt_pw1, 3, 64, pw1t, 32, 64);
    pack_bt_f16_kernel<<<CEILDIV(256 * 64, 256), 256, 0, stream>>>(pt_pw2, 64, 256, pw2t, 64, 256);
    pack_bt_f16_kernel<<<CEILDIV(64 * 256, 256), 256, 0, stream>>>(pt_aw1, 256, 64, aw1t, 256, 64);
    pack_bt_f16_kernel<<<CEILDIV(256 * 64, 256), 256, 0, stream>>>(pt_aw2, 64, 256, aw2t, 64, 256);
    pack_bt_f16_kernel<<<CEILDIV(16 * 256, 256), 256, 0, stream>>>(gn_s_w, 256, 4, gst, 256, 16);
    pack_bt_f16_kernel<<<CEILDIV(16 * 256, 256), 256, 0, stream>>>(gn_h_w, 256, 16, ght, 256, 16);
    pack_bt_f16_kernel<<<CEILDIV(512 * 288, 256), 256, 0, stream>>>(gn_o_w, 288, 512, got, 288, 512);
    pack_bt_f16_kernel<<<CEILDIV(1024 * 896, 256), 256, 0, stream>>>(fin_w, 896, 1024, fwt, 896, 1024);

    // ================= level 1: DynamicEdgeConv =================
    concat_xp_kernel<<<CEILDIV(n1, 256), 256, 0, stream>>>(x, pos, xp, n1);
    hipMemcpyAsync(o_pos, pos, (size_t)n1 * 3 * 4, hipMemcpyDeviceToDevice, stream);

    knn_kernel<11, true><<<n1 / 128, 128, 0, stream>>>(xp, n1, idx1);
    _Float16* m16 = AR4h;
    edge_gather1_kernel<<<CEILDIV(E1, 256), 256, 0, stream>>>(xp, idx1, m16, n1);

    _Float16* h1_16 = AR3h;            // [E1,64]
    _Float16* h2_16 = AR2h;            // [E1,128]
    float*    h3e   = AR0;             // [E1,128] f32
    wmma_gemm_kernel<1, 4, 1><<<dim3(E1 / 16, 1), 32, 0, stream>>>(m16, 32, w1t, 32, dec_b1, nullptr, 0, h1_16, 64);
    wmma_gemm_kernel<2, 4, 1><<<dim3(E1 / 16, 2), 32, 0, stream>>>(h1_16, 64, w2t, 64, dec_b2, nullptr, 0, h2_16, 128);
    wmma_gemm_kernel<4, 4, 0><<<dim3(E1 / 16, 2), 32, 0, stream>>>(h2_16, 128, w3t, 128, dec_b3, h3e, 128, nullptr, 0);

    float* t1 = (float*)AR2h;          // [n1,128] (h2_16 dead)
    max_over_k_kernel<<<CEILDIV(n1 * 128, 256), 256, 0, stream>>>(h3e, t1, n1, 128);
    ln_gelu_kernel<<<n1 / 8, 256, 0, stream>>>(t1, ln1_g, ln1_b, o_f1, n1, 128);

    // ---- downsample 1 ----
    score_kernel<<<n1 / 8, 256, 0, stream>>>(o_f1, pos, 128, 64, ds1_w1, ds1_b1, ds1_w2, ds1_b2, s1, n1);
    rank_select_kernel<<<n1 / 256, 256, 0, stream>>>(s1, n1, n2, sel1);
    gather_rows_kernel<<<CEILDIV(n2 * 128, 256), 256, 0, stream>>>(o_f1, sel1, x2, n2, 128);
    gather_rows_kernel<<<CEILDIV(n2 * 3, 256), 256, 0, stream>>>(pos, sel1, o_pos2, n2, 3);
    mean_rows_kernel<<<1, 256, 0, stream>>>(x2, comb + 0, n2, 128);   // pooled1
    pack_a_f16_kernel<<<CEILDIV(n2 * 128, 256), 256, 0, stream>>>(x2, n2, 128, x2h, n2, 128);

    // ================= level 2: PointTransformerConv =================
    knn_kernel<3, false><<<n2 / 128, 128, 0, stream>>>(o_pos2, n2, nbr);

    wmma_gemm_kernel<4, 4, 0><<<dim3(n2 / 16, 4), 32, 0, stream>>>(x2h, 128, plt, 128, nullptr, xl, 256, nullptr, 0);
    wmma_gemm_kernel<4, 4, 0><<<dim3(n2 / 16, 4), 32, 0, stream>>>(x2h, 128, pst, 128, nullptr, xs, 256, nullptr, 0);
    wmma_gemm_kernel<4, 4, 0><<<dim3(n2 / 16, 4), 32, 0, stream>>>(x2h, 128, pdt, 128, nullptr, xd, 256, nullptr, 0);

    _Float16* dpos16 = AR4h;                    // [E2,32]
    _Float16* hh16   = AR3h;                    // [E2,64]
    _Float16* hh2_16 = AR3h + (size_t)E2 * 64;  // [E2,64]
    float*    delta  = AR0;                     // [E2,256] f32
    _Float16* ain16  = AR2h;                    // [E2,256]
    float*    alpha  = AR1;                     // [E2,256] f32
    edge_dpos_kernel<<<CEILDIV(E2, 256), 256, 0, stream>>>(o_pos2, nbr, dpos16, n2);
    wmma_gemm_kernel<1, 4, 1><<<dim3(E2 / 16, 1), 32, 0, stream>>>(dpos16, 32, pw1t, 32, pt_pb1, nullptr, 0, hh16, 64);
    wmma_gemm_kernel<2, 4, 0><<<dim3(E2 / 16, 4), 32, 0, stream>>>(hh16, 64, pw2t, 64, pt_pb2, delta, 256, nullptr, 0);

    alpha_in_kernel<<<CEILDIV(E2 * 256, 256), 256, 0, stream>>>(xd, xs, delta, nbr, ain16, E2);
    wmma_gemm_kernel<8, 4, 1><<<dim3(E2 / 16, 1), 32, 0, stream>>>(ain16, 256, aw1t, 256, pt_ab1, nullptr, 0, hh2_16, 64);
    wmma_gemm_kernel<2, 4, 0><<<dim3(E2 / 16, 4), 32, 0, stream>>>(hh2_16, 64, aw2t, 64, pt_ab2, alpha, 256, nullptr, 0);

    // segment softmax over targets (amax/den/num contiguous -> one fill)
    fill_zero_kernel<<<CEILDIV(3 * n2 * 256, 256), 256, 0, stream>>>(amax, 3 * n2 * 256);
    scatter_max_kernel<<<CEILDIV(E2 * 256, 256), 256, 0, stream>>>(alpha, nbr, amax, E2);
    scatter_sum_kernel<<<CEILDIV(E2 * 256, 256), 256, 0, stream>>>(alpha, delta, xl, nbr, amax, den, num, E2);
    softmax_div_kernel<<<CEILDIV(n2 * 256, 256), 256, 0, stream>>>(num, den, out2, n2 * 256);
    ln_gelu_kernel<<<n2 / 8, 256, 0, stream>>>(out2, ln2_g, ln2_b, o_f2, n2, 256);

    // ---- downsample 2 ----
    score_kernel<<<n2 / 8, 256, 0, stream>>>(o_f2, o_pos2, 256, 128, ds2_w1, ds2_b1, ds2_w2, ds2_b2, s2, n2);
    rank_select_kernel<<<n2 / 256, 256, 0, stream>>>(s2, n2, n3, sel2);
    gather_rows_kernel<<<CEILDIV(n3 * 256, 256), 256, 0, stream>>>(o_f2, sel2, x3, n3, 256);
    gather_rows_kernel<<<CEILDIV(n3 * 3, 256), 256, 0, stream>>>(o_pos2, sel2, o_pos3, n3, 3);
    mean_rows_kernel<<<1, 256, 0, stream>>>(x3, comb + 128, n3, 256);  // pooled2
    pack_a_f16_kernel<<<CEILDIV(n3 * 256, 256), 256, 0, stream>>>(x3, n3, 256, x3h, n3, 256);

    // ================= level 3: GravNetConv =================
    wmma_gemm_kernel<8, 1, 0><<<dim3(n3 / 16, 1), 32, 0, stream>>>(x3h, 256, gst, 256, nullptr, s3p, 16, nullptr, 0);
    strip_s3_kernel<<<CEILDIV(n3 * 4, 256), 256, 0, stream>>>(s3p, gn_s_b, s3, n3);
    wmma_gemm_kernel<8, 1, 0><<<dim3(n3 / 16, 1), 32, 0, stream>>>(x3h, 256, ght, 256, gn_h_b, h3g, 16, nullptr, 0);
    knn_kernel<4, true><<<n3 / 128, 128, 0, stream>>>(s3, n3, idx3);
    gravnet_kernel<<<n3 / 8, 256, 0, stream>>>(x3, s3, h3g, idx3, gin16, n3);
    wmma_gemm_kernel<9, 4, 0><<<dim3(n3 / 16, 8), 32, 0, stream>>>(gin16, 288, got, 288, gn_o_b, o3, 512, nullptr, 0);
    ln_gelu_kernel<<<n3 / 8, 256, 0, stream>>>(o3, ln3_g, ln3_b, o_f3, n3, 512);
    mean_rows_kernel<<<2, 256, 0, stream>>>(o_f3, comb + 384, n3, 512);  // pooled3

    // ================= final readout =================
    pack_a_f16_kernel<<<CEILDIV(16 * 896, 256), 256, 0, stream>>>(comb, 1, 896, combh, 16, 896);
    wmma_gemm_kernel<28, 4, 0><<<dim3(1, 16), 32, 0, stream>>>(combh, 896, fwt, 896, fin_b, ftmp, 1024, nullptr, 0);
    ln_gelu_kernel<<<1, 32, 0, stream>>>(ftmp, fln_g, fln_b, o_out, 1, 1024);
}
